// EnhancedMultiheadAttention_6897717477672
// MI455X (gfx1250) — matrix operand order repaired and checked by reading, not compiled, not run on hardware
//
#include <hip/hip_runtime.h>
#include <hip/hip_bf16.h>

// ---------------------------------------------------------------------------
// Attention with 2-slope-ALiBi 2D bias, returning (attn_output, attn_weights).
// B=4, H=1, S=4096, E=head_dim=64. f32 in/out; bf16 WMMA internally.
//
// Roofline: 268MB attn_weights write ~12us @ 23.3 TB/s is the floor; bf16
// v_wmma_f32_16x16x32_bf16 compute (~17 GFLOP) is far below it, so scores are
// recomputed (stats sweep + emit sweep) instead of spilled -- K/V re-reads sit
// in the 192MB L2. Staging is software-pipelined: global loads for tile kt+1
// issue before the compute of tile kt; cvt+LDS stores land in the alternate
// buffer after compute; one barrier per tile.
// ---------------------------------------------------------------------------

typedef __attribute__((ext_vector_type(16))) __bf16          v16bf;
typedef __attribute__((ext_vector_type(16))) unsigned short  v16us;
typedef __attribute__((ext_vector_type(8)))  unsigned short  v8us;
typedef __attribute__((ext_vector_type(8)))  float           v8f;

#define S_LEN   4096
#define EMB     64
#define NBATCH  4
#define KTILE   32
#define NKT     (S_LEN / KTILE)          // 128 key tiles

#define LOG2E   1.44269504088896340736f
#define QKS     (LOG2E / 8.0f)           // 1/sqrt(64) folded with log2(e)
#define SL_L    (-0.5f * LOG2E)          // -(2^-1)   * log2e
#define SL_R    (-0.70710678118654752440f * LOG2E) // -(2^-0.5) * log2e

// LDS strides (elements). Multiples of 8 (16B) for aligned b128 frag loads,
// chosen so 16-lane accesses land on 16 distinct bank groups.
#define KSTR 72   // K tile  [32 x 64]  bf16
#define VSTR 40   // V^T tile[64 x 32]  bf16
#define PSTR 40   // P tile  [16 x 32]  bf16, per wave

__device__ __forceinline__ unsigned short f2bf(float f) {
  unsigned int u = __float_as_uint(f);
  u += 0x7fffu + ((u >> 16) & 1u);       // round-to-nearest-even
  return (unsigned short)(u >> 16);
}

__device__ __forceinline__ v8f wmma_bf16(v16us a, v16us b, v8f c) {
  return __builtin_amdgcn_wmma_f32_16x16x32_bf16(
      false, __builtin_bit_cast(v16bf, a),
      false, __builtin_bit_cast(v16bf, b),
      (short)0, c, false, false);
}

__device__ __forceinline__ v16us lds_v16(const unsigned short* p) {
  return *(const v16us*)p;
}

// reductions across the 16-lane half-wave that owns one N-column group
__device__ __forceinline__ float halfmax16(float v) {
  v = fmaxf(v, __shfl_xor(v, 1, 32));
  v = fmaxf(v, __shfl_xor(v, 2, 32));
  v = fmaxf(v, __shfl_xor(v, 4, 32));
  v = fmaxf(v, __shfl_xor(v, 8, 32));
  return v;
}
__device__ __forceinline__ float halfsum16(float v) {
  v += __shfl_xor(v, 1, 32);
  v += __shfl_xor(v, 2, 32);
  v += __shfl_xor(v, 4, 32);
  v += __shfl_xor(v, 8, 32);
  return v;
}

// ---- software-pipelined staging: global->regs, then regs->LDS(bf16) ----
__device__ __forceinline__ void load_tile_regs(const float* __restrict__ g,
                                               int kt, int tid, float4 r[4]) {
#pragma unroll
  for (int p = 0; p < 4; ++p) {
    int i   = tid + p * 128;             // 512 float4's cover 32x64 f32
    int row = i >> 4;
    int c4  = (i & 15) * 4;
    r[p] = *(const float4*)(g + (size_t)(kt * KTILE + row) * EMB + c4);
  }
}
__device__ __forceinline__ void store_K(const float4 r[4],
                                        unsigned short* sK, int tid) {
#pragma unroll
  for (int p = 0; p < 4; ++p) {
    int i   = tid + p * 128;
    int row = i >> 4;
    int c4  = (i & 15) * 4;
    unsigned short* d = &sK[row * KSTR + c4];
    d[0] = f2bf(r[p].x); d[1] = f2bf(r[p].y);
    d[2] = f2bf(r[p].z); d[3] = f2bf(r[p].w);
  }
}
__device__ __forceinline__ void store_VT(const float4 r[4],
                                         unsigned short* sVT, int tid) {
#pragma unroll
  for (int p = 0; p < 4; ++p) {
    int i   = tid + p * 128;
    int row = i >> 4;
    int c4  = (i & 15) * 4;
    sVT[(c4 + 0) * VSTR + row] = f2bf(r[p].x);
    sVT[(c4 + 1) * VSTR + row] = f2bf(r[p].y);
    sVT[(c4 + 2) * VSTR + row] = f2bf(r[p].z);
    sVT[(c4 + 3) * VSTR + row] = f2bf(r[p].w);
  }
}

// Scores (log2 domain, bias fused) for one 16q x 32k tile: two f32 C frags.
__device__ __forceinline__ void score_tile(const unsigned short* sK,
                                           v16us a0, v16us a1,
                                           int nn, int hi, int kt,
                                           const float* xr, const float* yr,
                                           int r0, v8f& z0, v8f& z1) {
  const int es = hi * 16;                // B frag: lanes>=16 hold e 16..31
  v16us b00 = lds_v16(&sK[nn * KSTR + es]);
  v16us b01 = lds_v16(&sK[nn * KSTR + 32 + es]);
  v16us b10 = lds_v16(&sK[(nn + 16) * KSTR + es]);
  v16us b11 = lds_v16(&sK[(nn + 16) * KSTR + 32 + es]);
  v8f zero = {};
  z0 = wmma_bf16(a0, b00, zero);
  z0 = wmma_bf16(a1, b01, z0);           // accumulate over e = 0..63
  z1 = wmma_bf16(a0, b10, zero);
  z1 = wmma_bf16(a1, b11, z1);

  const int c0 = kt * KTILE + nn;        // cols held by this lane
  const int c1 = c0 + 16;
  const float xc0 = (float)(c0 >> 6), yc0 = (float)(c0 & 63);
  const float xc1 = (float)(c1 >> 6), yc1 = (float)(c1 & 63);
#pragma unroll
  for (int j = 0; j < 8; ++j) {
    const int r = r0 + j;
    float d0 = fabsf(xr[j] - xc0) + fabsf(yr[j] - yc0);
    float d1 = fabsf(xr[j] - xc1) + fabsf(yr[j] - yc1);
    float s0 = (c0 > r) ? SL_R : SL_L;   // triu/tril slope select (diag: d=0)
    float s1 = (c1 > r) ? SL_R : SL_L;
    z0[j] = fmaf(z0[j], QKS, s0 * d0);
    z1[j] = fmaf(z1[j], QKS, s1 * d1);
  }
}

__global__ __launch_bounds__(128) void
attn_alibi2d_kernel(const float* __restrict__ Q, const float* __restrict__ K,
                    const float* __restrict__ V, float* __restrict__ Out) {
  __shared__ unsigned short sK [2][32 * KSTR];     // 2 x 4.6 KB
  __shared__ unsigned short sVT[2][64 * VSTR];     // 2 x 5.1 KB
  __shared__ unsigned short sP [4 * 16 * PSTR];    // 5.1 KB (per-wave slices)

  const int tid  = threadIdx.x;
  const int w    = tid >> 5;
  const int lane = tid & 31;
  const int nn   = lane & 15;
  const int hi   = lane >> 4;

  const int b    = blockIdx.x >> 6;      // 64 row-blocks per batch
  const int qw   = (blockIdx.x & 63) * 64 + w * 16;   // first q row of wave

  const float* gQ = Q + (size_t)b * S_LEN * EMB;
  const float* gK = K + (size_t)b * S_LEN * EMB;
  const float* gV = V + (size_t)b * S_LEN * EMB;
  float* gO = Out + (size_t)b * S_LEN * EMB;
  float* gW = Out + (size_t)NBATCH * S_LEN * EMB + (size_t)b * S_LEN * S_LEN;

  // ---- Q A-fragments (one-time, contiguous k = 16*hi + s order, same as K B-frags) ----
  const float* qrow = gQ + (size_t)(qw + nn) * EMB;
  const int qs = hi ? 16 : 0;
  v16us a0, a1;
#pragma unroll
  for (int t = 0; t < 8; ++t) {
    a0[t]     = f2bf(qrow[qs + t]);
    a0[8 + t] = f2bf(qrow[qs + 8 + t]);
    a1[t]     = f2bf(qrow[32 + qs + t]);
    a1[8 + t] = f2bf(qrow[32 + qs + 8 + t]);
  }

  // per-lane row coordinates for the 2D bias (C frag rows: j + 8*hi)
  const int r0 = qw + hi * 8;
  float xr[8], yr[8];
#pragma unroll
  for (int j = 0; j < 8; ++j) {
    int r = r0 + j;
    xr[j] = (float)(r >> 6);
    yr[j] = (float)(r & 63);
  }

  float m[8], l[8];
#pragma unroll
  for (int j = 0; j < 8; ++j) { m[j] = -1e30f; l[j] = 0.0f; }

  float4 kr[4], vr[4];

  // ================= sweep 1: online softmax stats =================
  load_tile_regs(gK, 0, tid, kr);
  store_K(kr, sK[0], tid);
  __syncthreads();
  for (int kt = 0; kt < NKT; ++kt) {
    const int cur  = kt & 1;
    const bool more = (kt + 1 < NKT);
    if (more) load_tile_regs(gK, kt + 1, tid, kr);   // overlap with compute

    v8f z0, z1;
    score_tile(sK[cur], a0, a1, nn, hi, kt, xr, yr, r0, z0, z1);
#pragma unroll
    for (int j = 0; j < 8; ++j) {
      float t  = halfmax16(fmaxf(z0[j], z1[j]));
      float mn = fmaxf(m[j], t);
      float e  = __builtin_amdgcn_exp2f(z0[j] - mn) +
                 __builtin_amdgcn_exp2f(z1[j] - mn);
      e = halfsum16(e);
      l[j] = l[j] * __builtin_amdgcn_exp2f(m[j] - mn) + e;
      m[j] = mn;
    }

    if (more) store_K(kr, sK[cur ^ 1], tid);
    __syncthreads();
  }

  float rl[8];
#pragma unroll
  for (int j = 0; j < 8; ++j) rl[j] = 1.0f / l[j];

  // ========= sweep 2: recompute scores, emit P, accumulate P.V =========
  v8f o0 = {}, o1 = {}, o2 = {}, o3 = {};
  unsigned short* pw = &sP[w * 16 * PSTR];

  load_tile_regs(gK, 0, tid, kr);
  load_tile_regs(gV, 0, tid, vr);
  store_K(kr, sK[0], tid);
  store_VT(vr, sVT[0], tid);
  __syncthreads();
  for (int kt = 0; kt < NKT; ++kt) {
    const int cur  = kt & 1;
    const bool more = (kt + 1 < NKT);
    if (more) {
      load_tile_regs(gK, kt + 1, tid, kr);           // overlap with compute
      load_tile_regs(gV, kt + 1, tid, vr);
    }

    v8f z0, z1;
    score_tile(sK[cur], a0, a1, nn, hi, kt, xr, yr, r0, z0, z1);
#pragma unroll
    for (int j = 0; j < 8; ++j) {
      float p0 = __builtin_amdgcn_exp2f(z0[j] - m[j]) * rl[j];
      float p1 = __builtin_amdgcn_exp2f(z1[j] - m[j]) * rl[j];
      // attn_weights (f32, coalesced 64B per half-wave)
      float* wr = gW + (size_t)(r0 + j) * S_LEN + kt * KTILE + nn;
      wr[0]  = p0;
      wr[16] = p1;
      // P -> LDS (bf16) for the cross-lane transpose into A-frag layout
      int lr = hi * 8 + j;
      pw[lr * PSTR + nn]      = f2bf(p0);
      pw[lr * PSTR + nn + 16] = f2bf(p1);
    }
    // P A-fragment: lane row nn, k chunks {0..15} / {16..31}
    const int ps = hi * 16;
    v8us plo = *(const v8us*)&pw[nn * PSTR + ps];
    v8us phi = *(const v8us*)&pw[nn * PSTR + ps + 8];
    v16us pa = __builtin_shufflevector(plo, phi, 0, 1, 2, 3, 4, 5, 6, 7,
                                       8, 9, 10, 11, 12, 13, 14, 15);
    const unsigned short* vt = sVT[cur];
    const int vs = hi * 16;              // B frag k range per half-wave
    o0 = wmma_bf16(pa, lds_v16(&vt[(0 * 16 + nn) * VSTR + vs]), o0);
    o1 = wmma_bf16(pa, lds_v16(&vt[(1 * 16 + nn) * VSTR + vs]), o1);
    o2 = wmma_bf16(pa, lds_v16(&vt[(2 * 16 + nn) * VSTR + vs]), o2);
    o3 = wmma_bf16(pa, lds_v16(&vt[(3 * 16 + nn) * VSTR + vs]), o3);

    if (more) {
      store_K(kr, sK[cur ^ 1], tid);
      store_VT(vr, sVT[cur ^ 1], tid);
    }
    __syncthreads();
  }

  // ---- epilogue: attn_output (already normalized) ----
#pragma unroll
  for (int j = 0; j < 8; ++j) {
    float* orow = gO + (size_t)(r0 + j) * EMB + nn;
    orow[0]  = o0[j];
    orow[16] = o1[j];
    orow[32] = o2[j];
    orow[48] = o3[j];
  }
}

extern "C" void kernel_launch(void* const* d_in, const int* in_sizes, int n_in,
                              void* d_out, int out_size, void* d_ws,
                              size_t ws_size, hipStream_t stream) {
  const float* Q = (const float*)d_in[0];
  const float* K = (const float*)d_in[1];
  const float* V = (const float*)d_in[2];
  float* O = (float*)d_out;
  attn_alibi2d_kernel<<<dim3(NBATCH * 64), dim3(128), 0, stream>>>(Q, K, V, O);
}
